// SMA_56719338111469
// MI455X (gfx1250) — compile-verified, matrix operand order checked by reading, not run on hardware
//
#include <hip/hip_runtime.h>
#include <hip/hip_bf16.h>

#define D_MODEL 768
#define N_HEADS 12
#define HD      64
#define B_DIM   2
#define L_DIM   2048
#define BL      (B_DIM * L_DIM)
#define K_SEL   46   // ceil(sqrt(2048))

typedef __attribute__((ext_vector_type(16))) __bf16 v16bf;
typedef __attribute__((ext_vector_type(8)))  float  v8f;
typedef __attribute__((ext_vector_type(4)))  unsigned v4u;

union Frag { v4u u[2]; v16bf v; };

// float -> bf16 RNE via bit ops
__device__ __forceinline__ __bf16 f2bf(float f) {
    unsigned u = __builtin_bit_cast(unsigned, f);
    unsigned r = u + 0x7FFFu + ((u >> 16) & 1u);
    unsigned short h = (unsigned short)(r >> 16);
    return __builtin_bit_cast(__bf16, h);
}

__device__ __forceinline__ v8f bwmma(v16bf a, v16bf b, v8f c) {
    return __builtin_amdgcn_wmma_f32_16x16x32_bf16(false, a, false, b, (short)0, c,
                                                   false, false);
}

// Async global->LDS 16B copy, GVS mode: saddr = uniform base, vaddr = per-lane
// byte offset, vdst = per-lane LDS byte address. Tracked by ASYNCcnt.
__device__ __forceinline__ void async_b128(unsigned lds_addr, const void* sbase,
                                           unsigned voff) {
    asm volatile("global_load_async_to_lds_b128 %0, %1, %2"
                 :: "v"(lds_addr), "v"(voff), "s"(sbase) : "memory");
}
#define WAIT_ASYNC(n) asm volatile("s_wait_asynccnt " #n ::: "memory")

__device__ __forceinline__ unsigned lds_off(const void* p) {
    return (unsigned)(size_t)p;   // flat LDS addr = addr[31:0]
}

// ---------------------------------------------------------------------------
// bf16 GEMM: C[M,N] = A[M,K] @ Wt[N,K]^T + bias.
// A and Wt both K-major bf16 in global; tiles stream via async-LDS, double
// buffered. Block 128 threads (4 waves); block tile 128x64; wave computes
// 32x64 -> 8 WMMAs per k-step from 12 b128 LDS loads.
// mode 0: fp32 row-major store. mode 1: QKV scatter store bf16 into
//   q,k: [B][H][L][HD]   vT: [B][H][HD][L]
// ---------------------------------------------------------------------------
__global__ __launch_bounds__(128)
void gemm_wmma_kernel(const __bf16* __restrict__ A, const __bf16* __restrict__ Wt,
                      const float* __restrict__ bias, float* __restrict__ out,
                      __bf16* __restrict__ qp, __bf16* __restrict__ kp,
                      __bf16* __restrict__ vp, int M, int N, int K, int mode) {
    __shared__ __align__(16) __bf16 at[2][128][32];
    __shared__ __align__(16) __bf16 bt[2][64][32];

    const int tid   = threadIdx.x;
    const int wave  = tid >> 5;
    const int lane  = tid & 31;
    const int nlane = lane & 15;
    const int khalf = lane >> 4;
    const int m0 = blockIdx.y * 128;
    const int n0 = blockIdx.x * 64;

    const __bf16* abase = A  + (size_t)m0 * K;
    const __bf16* bbase = Wt + (size_t)n0 * K;

    auto issue = [&](int buf, int k0) {
        unsigned lat = lds_off(&at[buf][0][0]);
        unsigned lbt = lds_off(&bt[buf][0][0]);
        #pragma unroll
        for (int ii = 0; ii < 4; ++ii) {            // A: 512 x 16B chunks
            int i   = tid + ii * 128;
            int row = i >> 2, c4 = i & 3;           // 4 chunks per 64B row
            unsigned va = (unsigned)(row * K + k0) * 2u + (unsigned)c4 * 16u;
            async_b128(lat + (unsigned)i * 16u, abase, va);
        }
        #pragma unroll
        for (int ii = 0; ii < 2; ++ii) {            // B: 256 x 16B chunks
            int i   = tid + ii * 128;
            int row = i >> 2, c4 = i & 3;
            unsigned va = (unsigned)(row * K + k0) * 2u + (unsigned)c4 * 16u;
            async_b128(lbt + (unsigned)i * 16u, bbase, va);
        }
    };

    v8f acc[2][4] = {};
    const int ksteps = K / 32;
    issue(0, 0);
    for (int s = 0; s < ksteps; ++s) {
        if (s + 1 < ksteps) { issue((s + 1) & 1, (s + 1) * 32); WAIT_ASYNC(6); }
        else                { WAIT_ASYNC(0); }
        __syncthreads();
        const int bi = s & 1;
        const int arow = wave * 32 + nlane;

        Frag af[2], bf[4];
        #pragma unroll
        for (int mt = 0; mt < 2; ++mt) {
            af[mt].u[0] = *(const v4u*)&at[bi][arow + mt * 16][khalf * 8];
            af[mt].u[1] = *(const v4u*)&at[bi][arow + mt * 16][16 + khalf * 8];
        }
        #pragma unroll
        for (int nt = 0; nt < 4; ++nt) {
            bf[nt].u[0] = *(const v4u*)&bt[bi][nt * 16 + nlane][khalf * 8];
            bf[nt].u[1] = *(const v4u*)&bt[bi][nt * 16 + nlane][16 + khalf * 8];
        }
        #pragma unroll
        for (int nt = 0; nt < 4; ++nt)
            #pragma unroll
            for (int mt = 0; mt < 2; ++mt)
                acc[mt][nt] = bwmma(af[mt].v, bf[nt].v, acc[mt][nt]);
        __syncthreads();
    }

    const int rhi = khalf;
    #pragma unroll
    for (int mt = 0; mt < 2; ++mt) {
        #pragma unroll
        for (int nt = 0; nt < 4; ++nt) {
            int n = n0 + nt * 16 + nlane;
            float bv = bias[n];
            #pragma unroll
            for (int r = 0; r < 8; ++r) {
                int m = m0 + wave * 32 + mt * 16 + r + 8 * rhi;
                float val = acc[mt][nt][r] + bv;
                if (mode == 0) {
                    out[(size_t)m * N + n] = val;
                } else {
                    int which = n / D_MODEL;
                    int d  = n - which * D_MODEL;
                    int h  = d >> 6;
                    int hd = d & 63;
                    int b  = m / L_DIM, t = m - b * L_DIM;
                    size_t bh = (size_t)(b * N_HEADS + h);
                    if (which == 0)      qp[(bh * L_DIM + t) * HD + hd] = f2bf(val);
                    else if (which == 1) kp[(bh * L_DIM + t) * HD + hd] = f2bf(val);
                    else                 vp[(bh * HD + hd) * L_DIM + t] = f2bf(val);
                }
            }
        }
    }
}

// ---------------------------------------------------------------------------
// Flash causal attention. Grid (L/64, B*H), block 128 (4 waves).
// q,k: bf16 [B][H][L][HD]; vT: bf16 [B][H][HD][L]; ctx out: bf16 [B][L][D].
// K and V tiles stream via async-LDS, double buffered.
// ---------------------------------------------------------------------------
__global__ __launch_bounds__(128)
void attn_wmma_kernel(const __bf16* __restrict__ q, const __bf16* __restrict__ kk,
                      const __bf16* __restrict__ vt, __bf16* __restrict__ ctx) {
    __shared__ __align__(16) __bf16 ktile[2][64][64];
    __shared__ __align__(16) __bf16 vtile[2][64][64];   // [hd][key]
    __shared__ __align__(16) __bf16 pls[4][16][64];     // per-wave P

    const int tid   = threadIdx.x;
    const int wave  = tid >> 5;
    const int lane  = tid & 31;
    const int nlane = lane & 15;
    const int khalf = lane >> 4;
    const int rhi   = khalf;

    const int bh = blockIdx.y;
    const int b  = bh / N_HEADS, h = bh % N_HEADS;
    const int qt = blockIdx.x;
    const float scale = 0.125f;   // 1/sqrt(HD), applied to f32 scores

    const __bf16* kbase_p = kk + (size_t)bh * L_DIM * HD;
    const __bf16* vbase_p = vt + (size_t)bh * HD * L_DIM;

    auto issue = [&](int buf, int kt_idx) {
        unsigned lk = lds_off(&ktile[buf][0][0]);
        unsigned lv = lds_off(&vtile[buf][0][0]);
        unsigned kgoff = (unsigned)(kt_idx * 64 * HD) * 2u;   // contiguous 8KB
        #pragma unroll
        for (int ii = 0; ii < 4; ++ii) {
            int i = tid + ii * 128;            // 512 x 16B chunks each
            async_b128(lk + (unsigned)i * 16u, kbase_p, kgoff + (unsigned)i * 16u);
            int row = i >> 3, c8 = i & 7;      // vT: row = hd, 8 chunks / 128B row
            unsigned va = (unsigned)(row * L_DIM + kt_idx * 64) * 2u + (unsigned)c8 * 16u;
            async_b128(lv + (unsigned)(row * 128 + c8 * 16), vbase_p, va);
        }
    };

    // Q fragments (2 x 16x32), straight b128 global loads
    const int qrow = qt * 64 + wave * 16 + nlane;
    const __bf16* qrp = q + ((size_t)bh * L_DIM + qrow) * HD;
    Frag qf[2];
    #pragma unroll
    for (int c = 0; c < 2; ++c) {
        qf[c].u[0] = *(const v4u*)(qrp + c * 32 + khalf * 8);
        qf[c].u[1] = *(const v4u*)(qrp + c * 32 + 16 + khalf * 8);
    }

    float mrun[8], lrun[8];
    #pragma unroll
    for (int r = 0; r < 8; ++r) { mrun[r] = -1e30f; lrun[r] = 0.f; }
    v8f cacc[4] = {};

    issue(0, 0);
    for (int kt_idx = 0; kt_idx <= qt; ++kt_idx) {
        if (kt_idx + 1 <= qt) { issue((kt_idx + 1) & 1, kt_idx + 1); WAIT_ASYNC(8); }
        else                  { WAIT_ASYNC(0); }
        __syncthreads();
        const int bi = kt_idx & 1;

        // S = Q @ K^T (4 key subtiles x 2 hd chunks), frags preloaded per tile
        v8f st[4];
        #pragma unroll
        for (int s = 0; s < 4; ++s) {
            Frag kf[2];
            #pragma unroll
            for (int c = 0; c < 2; ++c) {
                kf[c].u[0] = *(const v4u*)&ktile[bi][s * 16 + nlane][c * 32 + khalf * 8];
                kf[c].u[1] = *(const v4u*)&ktile[bi][s * 16 + nlane][c * 32 + 16 + khalf * 8];
            }
            v8f a = {};
            a = bwmma(qf[0].v, kf[0].v, a);
            a = bwmma(qf[1].v, kf[1].v, a);
            st[s] = a;
        }

        // scale + causal mask + online softmax
        const int qb = qt * 64 + wave * 16;
        #pragma unroll
        for (int r = 0; r < 8; ++r) {
            int qm = qb + r + 8 * rhi;
            float rowmax = -1e30f;
            #pragma unroll
            for (int s = 0; s < 4; ++s) {
                int kg = kt_idx * 64 + s * 16 + nlane;
                float sv = st[s][r] * scale;
                if (kg > qm) sv += -10000.0f;
                st[s][r] = sv;
                rowmax = fmaxf(rowmax, sv);
            }
            #pragma unroll
            for (int msk = 1; msk < 16; msk <<= 1)
                rowmax = fmaxf(rowmax, __shfl_xor(rowmax, msk, 32));
            float mnew  = fmaxf(mrun[r], rowmax);
            float alpha = __expf(mrun[r] - mnew);
            float lsum  = 0.f;
            int rloc = r + 8 * rhi;
            #pragma unroll
            for (int s = 0; s < 4; ++s) {
                float p = __expf(st[s][r] - mnew);
                lsum += p;
                pls[wave][rloc][s * 16 + nlane] = f2bf(p);
            }
            #pragma unroll
            for (int msk = 1; msk < 16; msk <<= 1)
                lsum += __shfl_xor(lsum, msk, 32);
            lrun[r] = lrun[r] * alpha + lsum;
            mrun[r] = mnew;
            #pragma unroll
            for (int t = 0; t < 4; ++t) cacc[t][r] *= alpha;
        }

        // ctx += P @ V, frags preloaded per tile
        Frag pf[2];
        #pragma unroll
        for (int c = 0; c < 2; ++c) {
            pf[c].u[0] = *(const v4u*)&pls[wave][nlane][c * 32 + khalf * 8];
            pf[c].u[1] = *(const v4u*)&pls[wave][nlane][c * 32 + 16 + khalf * 8];
        }
        #pragma unroll
        for (int ht = 0; ht < 4; ++ht) {
            Frag vf[2];
            #pragma unroll
            for (int c = 0; c < 2; ++c) {
                vf[c].u[0] = *(const v4u*)&vtile[bi][ht * 16 + nlane][c * 32 + khalf * 8];
                vf[c].u[1] = *(const v4u*)&vtile[bi][ht * 16 + nlane][c * 32 + 16 + khalf * 8];
            }
            cacc[ht] = bwmma(pf[0].v, vf[0].v, cacc[ht]);
            cacc[ht] = bwmma(pf[1].v, vf[1].v, cacc[ht]);
        }
        __syncthreads();
    }

    #pragma unroll
    for (int ht = 0; ht < 4; ++ht) {
        #pragma unroll
        for (int r = 0; r < 8; ++r) {
            int t = qt * 64 + wave * 16 + r + 8 * rhi;
            float val = cacc[ht][r] / lrun[r];
            ctx[((size_t)(b * L_DIM + t)) * D_MODEL + h * HD + ht * 16 + nlane] = f2bf(val);
        }
    }
}

// ---------------------------------------------------------------------------
// fp32 -> bf16 elementwise convert
// ---------------------------------------------------------------------------
__global__ __launch_bounds__(256)
void convert_kernel(const float* __restrict__ src, __bf16* __restrict__ dst, int n) {
    int i = blockIdx.x * 256 + threadIdx.x;
    if (i < n) dst[i] = f2bf(src[i]);
}

// fp32 W[K][N] -> bf16 Wt[N][K] (tiled transpose)
__global__ __launch_bounds__(256)
void convert_transpose_kernel(const float* __restrict__ W, __bf16* __restrict__ Wt,
                              int K, int N) {
    __shared__ float tile[32][33];
    const int n0 = blockIdx.x * 32, k0 = blockIdx.y * 32;
    const int r = threadIdx.x >> 5, c = threadIdx.x & 31;
    for (int rr = r; rr < 32; rr += 8)
        tile[rr][c] = W[(size_t)(k0 + rr) * N + n0 + c];
    __syncthreads();
    for (int rr = r; rr < 32; rr += 8)
        Wt[(size_t)(n0 + rr) * K + k0 + c] = f2bf(tile[c][rr]);
}

// ---------------------------------------------------------------------------
// selection[t] = softmax((x @ sel_w + sel_b)/T)[1]
// ---------------------------------------------------------------------------
__global__ __launch_bounds__(256)
void sel_kernel(const float* __restrict__ x, const float* __restrict__ sel_w,
                const float* __restrict__ sel_b, const float* __restrict__ temp,
                float* __restrict__ sel) {
    int t = blockIdx.x * 256 + threadIdx.x;
    if (t >= BL) return;
    const float* xr = x + (size_t)t * D_MODEL;
    float a0 = sel_b[0], a1 = sel_b[1];
    for (int i = 0; i < D_MODEL; ++i) {
        float xv = xr[i];
        a0 += xv * sel_w[2 * i];
        a1 += xv * sel_w[2 * i + 1];
    }
    float T = temp[0];
    a0 /= T; a1 /= T;
    float mx = fmaxf(a0, a1);
    float e0 = __expf(a0 - mx), e1 = __expf(a1 - mx);
    sel[t] = e1 / (e0 + e1);
}

// top-46 per batch, iterative argmax (distinct indices; lowest-index ties)
__global__ __launch_bounds__(256)
void topk_kernel(const float* __restrict__ sel, int* __restrict__ idx_out) {
    __shared__ float sv[L_DIM];
    __shared__ unsigned char taken[L_DIM];
    __shared__ float rv[256];
    __shared__ int   ri[256];
    const int b = blockIdx.x;
    for (int t = threadIdx.x; t < L_DIM; t += 256) {
        sv[t] = sel[(size_t)b * L_DIM + t];
        taken[t] = 0;
    }
    __syncthreads();
    for (int p = 0; p < K_SEL; ++p) {
        float best = -1e30f; int bi = L_DIM;
        for (int t = threadIdx.x; t < L_DIM; t += 256) {
            if (!taken[t] && (sv[t] > best || (sv[t] == best && t < bi))) {
                best = sv[t]; bi = t;
            }
        }
        rv[threadIdx.x] = best; ri[threadIdx.x] = bi;
        __syncthreads();
        for (int s = 128; s > 0; s >>= 1) {
            if (threadIdx.x < s) {
                float ov = rv[threadIdx.x + s]; int oi = ri[threadIdx.x + s];
                if (ov > rv[threadIdx.x] ||
                    (ov == rv[threadIdx.x] && oi < ri[threadIdx.x])) {
                    rv[threadIdx.x] = ov; ri[threadIdx.x] = oi;
                }
            }
            __syncthreads();
        }
        if (threadIdx.x == 0) {
            idx_out[b * K_SEL + p] = ri[0];
            taken[ri[0]] = 1;
        }
        __syncthreads();
    }
}

__global__ __launch_bounds__(256)
void copy_kernel(const float* __restrict__ x, float* __restrict__ y, int n) {
    int i = blockIdx.x * 256 + threadIdx.x;
    if (i < n) y[i] = x[i];
}

__global__ __launch_bounds__(256)
void scatter_kernel(const float* __restrict__ attn, const float* __restrict__ sel,
                    const int* __restrict__ idx, float* __restrict__ y) {
    int b = blockIdx.x / K_SEL;
    int j = blockIdx.x % K_SEL;
    int t = idx[b * K_SEL + j];
    float s = sel[(size_t)b * L_DIM + t];
    size_t off = ((size_t)b * L_DIM + t) * D_MODEL;
    for (int d = threadIdx.x; d < D_MODEL; d += 256)
        y[off + d] += attn[off + d] * s;
}

// ---------------------------------------------------------------------------
extern "C" void kernel_launch(void* const* d_in, const int* in_sizes, int n_in,
                              void* d_out, int out_size, void* d_ws, size_t ws_size,
                              hipStream_t stream) {
    const float* x      = (const float*)d_in[0];
    const float* Wqkv_w = (const float*)d_in[1];
    const float* Wqkv_b = (const float*)d_in[2];
    const float* sel_w  = (const float*)d_in[3];
    const float* sel_b  = (const float*)d_in[4];
    const float* out_w  = (const float*)d_in[5];
    const float* out_b  = (const float*)d_in[6];
    const float* temp   = (const float*)d_in[7];
    float* y = (float*)d_out;

    const size_t per = (size_t)BL * D_MODEL;   // 3,145,728
    __bf16* wsb = (__bf16*)d_ws;
    size_t o = 0;
    __bf16* xb    = wsb + o; o += per;
    __bf16* wqkvt = wsb + o; o += (size_t)3 * D_MODEL * D_MODEL;
    __bf16* outwt = wsb + o; o += (size_t)D_MODEL * D_MODEL;
    __bf16* q     = wsb + o; o += per;
    __bf16* k     = wsb + o; o += per;
    __bf16* vT    = wsb + o; o += per;
    __bf16* ctx   = wsb + o; o += per;
    float*  attn  = (float*)(wsb + o);
    float*  sel   = attn + per;
    int*    idx   = (int*)(sel + BL);

    // 0) one-time precision/layout conversion
    convert_kernel<<<(int)(per / 256), 256, 0, stream>>>(x, xb, (int)per);
    convert_transpose_kernel<<<dim3(3 * D_MODEL / 32, D_MODEL / 32), 256, 0, stream>>>(
        Wqkv_w, wqkvt, D_MODEL, 3 * D_MODEL);
    convert_transpose_kernel<<<dim3(D_MODEL / 32, D_MODEL / 32), 256, 0, stream>>>(
        out_w, outwt, D_MODEL, D_MODEL);

    // 1) QKV projection -> q/k/vT (bf16)
    gemm_wmma_kernel<<<dim3(3 * D_MODEL / 64, BL / 128), 128, 0, stream>>>(
        xb, wqkvt, Wqkv_b, nullptr, q, k, vT, BL, 3 * D_MODEL, D_MODEL, 1);
    // 2) causal flash attention -> ctx bf16 [B][L][D]
    attn_wmma_kernel<<<dim3(L_DIM / 64, B_DIM * N_HEADS), 128, 0, stream>>>(
        q, k, vT, ctx);
    // 3) output projection -> attn fp32 [B][L][D]
    gemm_wmma_kernel<<<dim3(D_MODEL / 64, BL / 128), 128, 0, stream>>>(
        ctx, outwt, out_b, attn, nullptr, nullptr, nullptr, BL, D_MODEL, D_MODEL, 0);
    // 4) selection gate
    sel_kernel<<<BL / 256, 256, 0, stream>>>(x, sel_w, sel_b, temp, sel);
    // 5) per-batch top-46
    topk_kernel<<<B_DIM, 256, 0, stream>>>(sel, idx);
    // 6) y = x ; scatter-add gated rows
    copy_kernel<<<(BL * D_MODEL) / 256, 256, 0, stream>>>(x, y, BL * D_MODEL);
    scatter_kernel<<<B_DIM * K_SEL, 256, 0, stream>>>(attn, sel, idx, y);
}
